// ChamferLoss_60146722013762
// MI455X (gfx1250) — matrix operand order repaired and checked by reading, not compiled, not run on hardware
//
#include <hip/hip_runtime.h>

// ---- CDNA5 (gfx1250) types ----
typedef __attribute__((ext_vector_type(16))) __bf16 v16bf;
typedef __attribute__((ext_vector_type(8)))  __bf16 bf16x8;
typedef __attribute__((ext_vector_type(4)))  __bf16 bf16x4;
typedef __attribute__((ext_vector_type(8)))  float  v8f;

#define B_   16
#define N_   4096
#define M_   4096
#define D_   64
#define BN   128          // rows (N) per block
#define BM   128          // cols (M) per chunk
#define LDA  72           // bf16 row stride in LDS (64 padded -> 72, avoids bank conflicts)
#define NCHUNK (M_ / BM)

// Order-preserving float <-> uint mapping (monotonic), so we can use
// integer atomic-min (ds_min_u32 / global_atomic_min_u32) for float mins.
__device__ __forceinline__ unsigned encf(float f) {
    int i = __float_as_int(f);
    return (unsigned)(i ^ ((i >> 31) | 0x80000000));
}
__device__ __forceinline__ float decf(unsigned u) {
    int i = (int)((u & 0x80000000u) ? (u ^ 0x80000000u) : ~u);
    return __int_as_float(i);
}

__global__ __launch_bounds__(256)
void chamfer_init_kernel(unsigned* __restrict__ colmin, float* __restrict__ rowsum, int n) {
    int g = blockIdx.x * blockDim.x + threadIdx.x;
    if (g == 0) rowsum[0] = 0.0f;
    if (g < n) colmin[g] = 0xFFFFFFFFu;   // encoded "max"
}

__global__ __launch_bounds__(256)
void chamfer_main_kernel(const float* __restrict__ pred, const float* __restrict__ tgt,
                         unsigned* __restrict__ colmin_g, float* __restrict__ rowsum_g)
{
    __shared__ __bf16 sAh[BN * LDA];
    __shared__ __bf16 sAl[BN * LDA];
    __shared__ __bf16 sBh[BM * LDA];
    __shared__ __bf16 sBl[BM * LDA];
    __shared__ float    sSq1[BN];
    __shared__ float    sSq2[BM];
    __shared__ unsigned sColMin[BM];

    const int b    = blockIdx.y;
    const int n0   = blockIdx.x * BN;
    const int tid  = threadIdx.x;
    const int lane = tid & 31;
    const int wv   = tid >> 5;       // wave 0..7 -> rows [16*wv, 16*wv+16)
    const int l16  = lane & 15;
    const int hf   = lane >> 4;      // wave half (WMMA K-interleave select)

    const float* predB = pred + (size_t)b * N_ * D_;
    const float* tgtB  = tgt  + (size_t)b * M_ * D_;

    // ---- stage pred strip (BN x D fp32) as hi/lo bf16 in LDS ----
    #pragma unroll
    for (int i = 0; i < (BN * D_) / (4 * 256); ++i) {   // 8 float4 per thread
        int e = (i * 256 + tid) * 4;
        int r = e / D_, c = e % D_;
        float4 v = *(const float4*)(predB + (size_t)(n0 + r) * D_ + c);
        bf16x4 h, l;
        h.x = (__bf16)v.x; l.x = (__bf16)(v.x - (float)h.x);
        h.y = (__bf16)v.y; l.y = (__bf16)(v.y - (float)h.y);
        h.z = (__bf16)v.z; l.z = (__bf16)(v.z - (float)h.z);
        h.w = (__bf16)v.w; l.w = (__bf16)(v.w - (float)h.w);
        *(bf16x4*)&sAh[r * LDA + c] = h;
        *(bf16x4*)&sAl[r * LDA + c] = l;
    }
    if (tid < BN) {   // ||x||^2 per row (fp32, exact)
        const float* rp = predB + (size_t)(n0 + tid) * D_;
        float s = 0.f;
        #pragma unroll
        for (int c = 0; c < D_; c += 4) {
            float4 v = *(const float4*)(rp + c);
            s += v.x * v.x + v.y * v.y + v.z * v.z + v.w * v.w;
        }
        sSq1[tid] = s;
    }

    // per-lane running row-min: rm[j] tracks row (16*wv + 8*hf + j)
    float rm[8];
    #pragma unroll
    for (int j = 0; j < 8; ++j) rm[j] = __builtin_inff();

    union Frag { v16bf v; bf16x8 h[2]; };

    for (int ch = 0; ch < NCHUNK; ++ch) {
        const int m0 = ch * BM;
        __syncthreads();   // previous chunk's B-tile fully consumed

        // stage target chunk (BM x D fp32) as hi/lo bf16
        #pragma unroll
        for (int i = 0; i < (BM * D_) / (4 * 256); ++i) {   // 8 float4 per thread
            int e = (i * 256 + tid) * 4;
            int r = e / D_, c = e % D_;
            float4 v = *(const float4*)(tgtB + (size_t)(m0 + r) * D_ + c);
            bf16x4 h, l;
            h.x = (__bf16)v.x; l.x = (__bf16)(v.x - (float)h.x);
        	h.y = (__bf16)v.y; l.y = (__bf16)(v.y - (float)h.y);
        	h.z = (__bf16)v.z; l.z = (__bf16)(v.z - (float)h.z);
        	h.w = (__bf16)v.w; l.w = (__bf16)(v.w - (float)h.w);
            *(bf16x4*)&sBh[r * LDA + c] = h;
            *(bf16x4*)&sBl[r * LDA + c] = l;
        }
        if (tid < BM) {    // ||y||^2 per chunk row + reset per-chunk col-min
            const float* rp = tgtB + (size_t)(m0 + tid) * D_;
            float s = 0.f;
            #pragma unroll
            for (int c = 0; c < D_; c += 4) {
                float4 v = *(const float4*)(rp + c);
                s += v.x * v.x + v.y * v.y + v.z * v.z + v.w * v.w;
            }
            sSq2[tid] = s;
            sColMin[tid] = 0xFFFFFFFFu;
        }
        if (ch + 1 < NCHUNK)   // global_prefetch_b8 of next chunk
            __builtin_prefetch(tgtB + (size_t)(m0 + BM) * D_ + tid * 16, 0, 0);
        __syncthreads();

        const int rowBase = wv * 16;

        // ---- hoist A fragments (invariant across column tiles): 4 frags, 32 VGPRs ----
        Frag ah[2], al[2];
        #pragma unroll
        for (int kk = 0; kk < D_ / 32; ++kk) {
            const int ar = rowBase + l16;
            const int ak = kk * 32 + 8 * hf;
            ah[kk].h[0] = *(const bf16x8*)&sAh[ar * LDA + ak];
            ah[kk].h[1] = *(const bf16x8*)&sAh[ar * LDA + ak + 16];
            al[kk].h[0] = *(const bf16x8*)&sAl[ar * LDA + ak];
            al[kk].h[1] = *(const bf16x8*)&sAl[ar * LDA + ak + 16];
        }

        #pragma unroll
        for (int t = 0; t < BM / 16; ++t) {      // 8 column tiles of 16
            v8f acc = {0.f, 0.f, 0.f, 0.f, 0.f, 0.f, 0.f, 0.f};
            #pragma unroll
            for (int kk = 0; kk < D_ / 32; ++kk) {   // K in chunks of 32
                const int k0 = kk * 32;
                Frag bh, bl;
                // B (32x16): lane=col, half0 -> K {k0..k0+7, k0+16..k0+23}, half1 -> K {k0+8..k0+15, k0+24..k0+31}
                const int br = t * 16 + l16;
                const int bk = k0 + 8 * hf;
                bh.h[0] = *(const bf16x8*)&sBh[br * LDA + bk];
                bh.h[1] = *(const bf16x8*)&sBh[br * LDA + bk + 16];
                bl.h[0] = *(const bf16x8*)&sBl[br * LDA + bk];
                bl.h[1] = *(const bf16x8*)&sBl[br * LDA + bk + 16];
                // fp32-accurate cross via compensated bf16: hi*hi + hi*lo + lo*hi
                acc = __builtin_amdgcn_wmma_f32_16x16x32_bf16(false, ah[kk].v, false, bh.v,
                                                              (short)0, acc, false, false);
                acc = __builtin_amdgcn_wmma_f32_16x16x32_bf16(false, ah[kk].v, false, bl.v,
                                                              (short)0, acc, false, false);
                acc = __builtin_amdgcn_wmma_f32_16x16x32_bf16(false, al[kk].v, false, bh.v,
                                                              (short)0, acc, false, false);
            }
            // dist = ||x||^2 + ||y||^2 - 2*cross ; fused row-min & col-min
            const float s2 = sSq2[t * 16 + l16];
            float cmin = __builtin_inff();
            #pragma unroll
            for (int j = 0; j < 8; ++j) {
                const float s1 = sSq1[rowBase + 8 * hf + j];
                const float d  = s1 + s2 - 2.0f * acc[j];
                rm[j] = fminf(rm[j], d);
                cmin  = fminf(cmin, d);
            }
            cmin = fminf(cmin, __shfl_xor(cmin, 16, 32));   // combine wave halves
            if (hf == 0) atomicMin(&sColMin[t * 16 + l16], encf(cmin));  // ds_min_u32
        }
        __syncthreads();
        if (tid < BM)
            atomicMin(&colmin_g[(size_t)b * M_ + m0 + tid], sColMin[tid]);
    }

    // ---- row-min wrap-up: cross-lane reduce within 16-lane groups, sum, one atomicAdd/wave ----
    float s = 0.f;
    #pragma unroll
    for (int j = 0; j < 8; ++j) {
        float v = rm[j];
        v = fminf(v, __shfl_xor(v, 1, 32));
        v = fminf(v, __shfl_xor(v, 2, 32));
        v = fminf(v, __shfl_xor(v, 4, 32));
        v = fminf(v, __shfl_xor(v, 8, 32));
        s += v;
    }
    s += __shfl_xor(s, 16, 32);   // rows 0-7 half + rows 8-15 half
    if (lane == 0) atomicAdd(rowsum_g, s);
}

__global__ __launch_bounds__(256)
void chamfer_finalize_kernel(const unsigned* __restrict__ colmin,
                             const float* __restrict__ rowsum,
                             float* __restrict__ out)
{
    __shared__ float red[256];
    float s = 0.f;
    for (int i = threadIdx.x; i < B_ * M_; i += 256) s += decf(colmin[i]);
    red[threadIdx.x] = s;
    __syncthreads();
    for (int st = 128; st > 0; st >>= 1) {
        if (threadIdx.x < st) red[threadIdx.x] += red[threadIdx.x + st];
        __syncthreads();
    }
    if (threadIdx.x == 0)
        out[0] = rowsum[0] / (float)(B_ * N_) + red[0] / (float)(B_ * M_);
}

extern "C" void kernel_launch(void* const* d_in, const int* in_sizes, int n_in,
                              void* d_out, int out_size, void* d_ws, size_t ws_size,
                              hipStream_t stream)
{
    const float* pred = (const float*)d_in[0];   // (B, N, D) fp32
    const float* tgt  = (const float*)d_in[1];   // (B, M, D) fp32
    float* out = (float*)d_out;                  // scalar

    float*    rowsum = (float*)d_ws;                         // 1 float @ offset 0
    unsigned* colmin = (unsigned*)((char*)d_ws + 256);       // B*M encoded mins

    chamfer_init_kernel<<<(B_ * M_ + 255) / 256, 256, 0, stream>>>(colmin, rowsum, B_ * M_);

    dim3 grid(N_ / BN, B_);
    chamfer_main_kernel<<<grid, 256, 0, stream>>>(pred, tgt, colmin, rowsum);

    chamfer_finalize_kernel<<<1, 256, 0, stream>>>(colmin, rowsum, out);
}
